// TransConvLayer_360777253132
// MI455X (gfx1250) — matrix operand order repaired and checked by reading, not compiled, not run on hardware
//
#include <hip/hip_runtime.h>

#define N_NODES 131072
#define IN_CH   512
#define HD      512     // HEADS * OUT_CH
#define D_OUT   128
#define HEADS   4

typedef __bf16 bf16_t;
typedef __bf16 v16bf __attribute__((ext_vector_type(16)));
typedef float  v8f   __attribute__((ext_vector_type(8)));
typedef unsigned int u32x4 __attribute__((ext_vector_type(4)));
typedef float  f32x4 __attribute__((ext_vector_type(4)));

union FragU  { v16bf v; u32x4 q[2]; };
union BfPack { v16bf v; bf16_t e[16]; };
union U128B  { u32x4 q; bf16_t b[8]; };
union U32P   { unsigned int u; bf16_t b[2]; };

// Load A/B fragment as two contiguous 16B chunks (possibly discontiguous halves).
__device__ __forceinline__ v16bf frag_ld2(const bf16_t* p0, const bf16_t* p1) {
    FragU f;
    f.q[0] = *(const u32x4*)p0;
    f.q[1] = *(const u32x4*)p1;
    return f.v;
}
// Load fragment as one contiguous 32B chunk.
__device__ __forceinline__ v16bf frag_ld32(const bf16_t* p) {
    FragU f;
    const u32x4* q = (const u32x4*)p;
    f.q[0] = q[0];
    f.q[1] = q[1];
    return f.v;
}
// Load 2x8 f32 and convert to a bf16 fragment in registers.
__device__ __forceinline__ v16bf frag_cvt(const float* p0, const float* p1) {
    BfPack r;
#pragma unroll
    for (int i = 0; i < 8; ++i) r.e[i]     = (bf16_t)p0[i];
#pragma unroll
    for (int i = 0; i < 8; ++i) r.e[8 + i] = (bf16_t)p1[i];
    return r.v;
}
// Pack two f32 into one dword of bf16 pair (v_cvt_pk_bf16_f32).
__device__ __forceinline__ unsigned int pack_bf16(float a, float b) {
    U32P p;
    p.b[0] = (bf16_t)a;
    p.b[1] = (bf16_t)b;
    return p.u;
}
__device__ __forceinline__ unsigned int pack2(bf16_t a, bf16_t b) {
    U32P p;
    p.b[0] = a;
    p.b[1] = b;
    return p.u;
}

__device__ __forceinline__ v8f wmma_bf16(v16bf a, v16bf b, v8f c) {
    return __builtin_amdgcn_wmma_f32_16x16x32_bf16(false, a, false, b, (short)0, c,
                                                   false, false);
}

// ---------------------------------------------------------------------------
// Kernel Z: zero small accumulators (kvs + ks_sum + sumsq, contiguous in ws)
// ---------------------------------------------------------------------------
__global__ void zero_kernel(float* __restrict__ p, int n) {
    int i = blockIdx.x * 256 + threadIdx.x;
    if (i < n) p[i] = 0.0f;
}

// ---------------------------------------------------------------------------
// Kernel A: fused QKV projection GEMM  (grid.z: 0=Q, 1=K, 2=V)
//   C[128x128] tile per workgroup; W tile staged transposed (bf16) in a
//   double-buffered LDS tile; X fed directly from global with f32->bf16
//   conversion in registers (A-operand layout is two contiguous 16B runs per
//   lane, so no LDS round-trip is needed for A).
//   Epilogue: bias add, sum-of-squares (Q,K), ks column sums (K),
//   stores Q/K as bf16, V as f32.
// ---------------------------------------------------------------------------
__global__ __launch_bounds__(256) void qkv_proj_kernel(
    const float* __restrict__ Xq, const float* __restrict__ Xs,
    const float* __restrict__ Wq, const float* __restrict__ bq,
    const float* __restrict__ Wk, const float* __restrict__ bk,
    const float* __restrict__ Wv, const float* __restrict__ bv,
    bf16_t* __restrict__ Qb, bf16_t* __restrict__ Kb, float* __restrict__ Vf,
    float* __restrict__ sumsq, float* __restrict__ ks_sum)
{
    const int mode = blockIdx.z;
    const float* X    = (mode == 0) ? Xq : Xs;
    const float* W    = (mode == 0) ? Wq : ((mode == 1) ? Wk : Wv);
    const float* bias = (mode == 0) ? bq : ((mode == 1) ? bk : bv);

    const int m0 = blockIdx.y * 128;
    const int n0 = blockIdx.x * 128;

    __shared__ bf16_t Wt[2][128][40];  // W tile transposed: Wt[buf][n][k], BK=32
    __shared__ float  ksPart[128];

    const int tid  = threadIdx.x;
    const int wave = tid >> 5;
    const int lane = tid & 31;
    const int wm   = wave & 3;         // 4 wave-rows  -> 32 rows each
    const int wn   = wave >> 2;        // 2 wave-cols  -> 64 cols each
    const int lm   = lane & 15;
    const int hi   = lane >> 4;
    const int khalf = hi ? 8 : 0;

    // Stage W[k0..k0+32)[n0..n0+128) transposed as bf16, processing k-row
    // pairs so stores are packed b32.  Load phase first (overlapped loads),
    // then convert+store phase.
    auto stage = [&](int buf, int k0) {
        f32x4 w0[2], w1[2];
#pragma unroll
        for (int i = 0; i < 2; ++i) {
            int e  = tid + i * 256;          // pair-task id: 16 kp * 32 chunks
            int kp = e >> 5;                 // 0..15 (k row pair)
            int n4 = (e & 31) << 2;          // 0..124
            const float* p = W + (size_t)(k0 + kp * 2) * HD + n0 + n4;
            w0[i] = *(const f32x4*)p;
            w1[i] = *(const f32x4*)(p + HD);
        }
#pragma unroll
        for (int i = 0; i < 2; ++i) {
            int e  = tid + i * 256;
            int kp = e >> 5;
            int n4 = (e & 31) << 2;
#pragma unroll
            for (int j = 0; j < 4; ++j)
                *(unsigned int*)&Wt[buf][n4 + j][kp * 2] =
                    pack_bf16(w0[i][j], w1[i][j]);
        }
    };

    const v8f zero8 = {0.f, 0.f, 0.f, 0.f, 0.f, 0.f, 0.f, 0.f};
    v8f acc[2][4];
#pragma unroll
    for (int t = 0; t < 2; ++t)
#pragma unroll
        for (int u = 0; u < 4; ++u) acc[t][u] = zero8;

    auto compute = [&](int buf, int k0) {
        // A fragments straight from global f32 (per-lane two 16B-equiv runs).
        v16bf afr[2];
#pragma unroll
        for (int t = 0; t < 2; ++t) {
            const float* rp =
                X + (size_t)(m0 + wm * 32 + t * 16 + lm) * IN_CH + k0;
            afr[t] = frag_cvt(rp + khalf, rp + 16 + khalf);
        }
#pragma unroll
        for (int u = 0; u < 4; ++u) {
            v16bf bfr = frag_ld2(&Wt[buf][wn * 64 + u * 16 + lm][khalf],
                                 &Wt[buf][wn * 64 + u * 16 + lm][16 + khalf]);
#pragma unroll
            for (int t = 0; t < 2; ++t)
                acc[t][u] = wmma_bf16(afr[t], bfr, acc[t][u]);
        }
    };

    stage(0, 0);
    __syncthreads();

    for (int k0 = 0; k0 < IN_CH - 32; k0 += 32) {
        const int buf = (k0 >> 5) & 1;
        stage(buf ^ 1, k0 + 32);              // overlaps with compute below
        if (k0 + 64 < IN_CH) {                // prefetch 2 tiles ahead
            const float* pf =
                W + (size_t)(k0 + 64 + (tid >> 3)) * HD + n0 + ((tid & 7) << 4);
            __builtin_prefetch(pf, 0, 1);
        }
        compute(buf, k0);
        __syncthreads();
    }
    compute(((IN_CH - 32) >> 5) & 1, IN_CH - 32);

    __syncthreads();
    if (tid < 128) ksPart[tid] = 0.0f;
    __syncthreads();

    float sq = 0.0f;
#pragma unroll
    for (int t = 0; t < 2; ++t) {
#pragma unroll
        for (int u = 0; u < 4; ++u) {
            const int cloc = wn * 64 + u * 16 + lm;
            const int col  = n0 + cloc;
            const float b  = bias[col];
            float csum = 0.0f;
#pragma unroll
            for (int j = 0; j < 8; ++j) {
                const int row = m0 + wm * 32 + t * 16 + j + hi * 8;
                const float c = acc[t][u][j] + b;
                const size_t idx = (size_t)row * HD + col;
                if (mode == 0) {
                    Qb[idx] = (bf16_t)c;
                    sq += c * c;
                } else if (mode == 1) {
                    Kb[idx] = (bf16_t)c;
                    sq += c * c;
                    csum += c;
                } else {
                    Vf[idx] = c;
                }
            }
            if (mode == 1) atomicAdd(&ksPart[cloc], csum);
        }
    }

    if (mode < 2) {
#pragma unroll
        for (int off = 16; off > 0; off >>= 1) sq += __shfl_xor(sq, off, 32);
        if (lane == 0) atomicAdd(&sumsq[mode], sq);
    }
    __syncthreads();
    if (mode == 1 && tid < 128) atomicAdd(&ks_sum[n0 + tid], ksPart[tid]);
}

// ---------------------------------------------------------------------------
// Kernel B: kvs[h] += K[:,h]^T @ V[:,h]  (split-K over node chunks)
//   K (bf16, b128 loads) and V (f32->bf16) staged transposed into a
//   double-buffered LDS tile, row-pair packing for b32 LDS stores.
// ---------------------------------------------------------------------------
__global__ __launch_bounds__(256) void kv_outer_kernel(
    const bf16_t* __restrict__ Kb, const float* __restrict__ Vf,
    float* __restrict__ kvs)
{
    const int h      = blockIdx.y;
    const int l0base = blockIdx.x * 2048;      // 64 chunks of 2048 rows

    __shared__ bf16_t Kt[2][128][40];  // Kt[b][m][l] = K[l0+l, h*128+m]
    __shared__ bf16_t Vt[2][128][40];  // Vt[b][d][l] = V[l0+l, h*128+d]

    const int tid  = threadIdx.x;
    const int wave = tid >> 5;
    const int lane = tid & 31;
    const int lm   = lane & 15;
    const int hi   = lane >> 4;
    const int khalf = hi ? 8 : 0;

    auto stage = [&](int buf, int l0) {
        // K tile: 32 rows x 128 cols bf16; one row-pair task per thread.
        {
            const int rp = tid >> 4;           // 0..15 (row pair)
            const int c8 = (tid & 15) << 3;    // 0..120
            const bf16_t* p = Kb + (size_t)(l0base + l0 + rp * 2) * HD +
                              h * D_OUT + c8;
            U128B t0, t1;
            t0.q = *(const u32x4*)p;
            t1.q = *(const u32x4*)(p + HD);
#pragma unroll
            for (int j = 0; j < 8; ++j)
                *(unsigned int*)&Kt[buf][c8 + j][rp * 2] = pack2(t0.b[j], t1.b[j]);
        }
        // V tile: 32 rows x 128 cols f32; two row-pair tasks per thread.
        f32x4 v0[2], v1[2];
#pragma unroll
        for (int i = 0; i < 2; ++i) {
            int e  = tid + i * 256;            // 16 rp * 32 chunks
            int rp = e >> 5;
            int c4 = (e & 31) << 2;
            const float* p = Vf + (size_t)(l0base + l0 + rp * 2) * HD +
                             h * D_OUT + c4;
            v0[i] = *(const f32x4*)p;
            v1[i] = *(const f32x4*)(p + HD);
        }
#pragma unroll
        for (int i = 0; i < 2; ++i) {
            int e  = tid + i * 256;
            int rp = e >> 5;
            int c4 = (e & 31) << 2;
#pragma unroll
            for (int j = 0; j < 4; ++j)
                *(unsigned int*)&Vt[buf][c4 + j][rp * 2] =
                    pack_bf16(v0[i][j], v1[i][j]);
        }
    };

    const v8f zero8 = {0.f, 0.f, 0.f, 0.f, 0.f, 0.f, 0.f, 0.f};
    v8f acc[8];
#pragma unroll
    for (int u = 0; u < 8; ++u) acc[u] = zero8;

    auto compute = [&](int buf) {
        const bf16_t* ap = &Kt[buf][wave * 16 + lm][0];
        v16bf afr = frag_ld2(ap + khalf, ap + 16 + khalf);
#pragma unroll
        for (int u = 0; u < 8; ++u) {
            v16bf bfr = frag_ld2(&Vt[buf][u * 16 + lm][khalf],
                                 &Vt[buf][u * 16 + lm][16 + khalf]);
            acc[u] = wmma_bf16(afr, bfr, acc[u]);
        }
    };

    stage(0, 0);
    __syncthreads();

    for (int l0 = 0; l0 < 2048 - 32; l0 += 32) {
        const int buf = (l0 >> 5) & 1;
        stage(buf ^ 1, l0 + 32);
        if (l0 + 64 < 2048) {
            const size_t prow = (size_t)(l0base + l0 + 64 + (tid >> 3)) * HD;
            __builtin_prefetch(Kb + prow + h * D_OUT + ((tid & 7) << 4), 0, 1);
            __builtin_prefetch(Vf + prow + h * D_OUT + ((tid & 7) << 4), 0, 1);
        }
        compute(buf);
        __syncthreads();
    }
    compute(((2048 - 32) >> 5) & 1);

    float* basep = kvs + (size_t)h * D_OUT * D_OUT;
#pragma unroll
    for (int u = 0; u < 8; ++u) {
#pragma unroll
        for (int j = 0; j < 8; ++j) {
            const int m = wave * 16 + j + hi * 8;
            const int d = u * 16 + lm;
            atomicAdd(&basep[m * D_OUT + d], acc[u][j]);
        }
    }
}

// ---------------------------------------------------------------------------
// Kernel D: kvs f32 -> bf16
// ---------------------------------------------------------------------------
__global__ void cvt_kvs_kernel(const float* __restrict__ kvs,
                               bf16_t* __restrict__ kvsb) {
    int i = blockIdx.x * 256 + threadIdx.x;
    kvsb[i] = (bf16_t)kvs[i];
}

// ---------------------------------------------------------------------------
// Kernel C: out[n,d] = mean_h [ (Q@kvs[h])/s + N*V ] / [ (Q . ks_sum[h])/s + N ]
//   s = ||q|| * ||k||.  kvs[h] staged transposed (bf16, b128 loads, row-pair
//   packed b32 LDS stores), Q A-frags straight from global bf16.
// ---------------------------------------------------------------------------
__global__ __launch_bounds__(256) void attn_out_kernel(
    const bf16_t* __restrict__ Qb, const float* __restrict__ Vf,
    const bf16_t* __restrict__ kvsb, const float* __restrict__ ks_sum,
    const float* __restrict__ sumsq, float* __restrict__ out)
{
    const int m0 = blockIdx.x * 128;

    __shared__ bf16_t Bt[128][136];  // Bt[d][m] = kvs[h][m][d]
    __shared__ float  ksl[128];
    __shared__ float  normz[128];

    const int tid  = threadIdx.x;
    const int wave = tid >> 5;
    const int lane = tid & 31;
    const int lm   = lane & 15;
    const int hi   = lane >> 4;
    const int khalf = hi ? 8 : 0;

    const float s     = sqrtf(sumsq[0]) * sqrtf(sumsq[1]);
    const float inv_s = 1.0f / s;
    const float Nf    = (float)N_NODES;

    float oacc[8][8];
#pragma unroll
    for (int u = 0; u < 8; ++u)
#pragma unroll
        for (int j = 0; j < 8; ++j) oacc[u][j] = 0.0f;

    const v8f zero8 = {0.f, 0.f, 0.f, 0.f, 0.f, 0.f, 0.f, 0.f};

    for (int h = 0; h < HEADS; ++h) {
        __syncthreads();
        // Stage kvs[h] transposed: 128x128 bf16, row-pair packed stores.
        const bf16_t* kb = kvsb + (size_t)h * D_OUT * D_OUT;
        U128B t0[4], t1[4];
#pragma unroll
        for (int i = 0; i < 4; ++i) {
            int e  = tid + i * 256;       // 64 row-pairs * 16 chunks
            int mp = e >> 4;
            int d8 = (e & 15) << 3;
            const bf16_t* p = kb + (size_t)(mp * 2) * D_OUT + d8;
            t0[i].q = *(const u32x4*)p;
            t1[i].q = *(const u32x4*)(p + D_OUT);
        }
#pragma unroll
        for (int i = 0; i < 4; ++i) {
            int e  = tid + i * 256;
            int mp = e >> 4;
            int d8 = (e & 15) << 3;
#pragma unroll
            for (int j = 0; j < 8; ++j)
                *(unsigned int*)&Bt[d8 + j][mp * 2] =
                    pack2(t0[i].b[j], t1[i].b[j]);
        }
        if (tid < 128) ksl[tid] = ks_sum[h * D_OUT + tid];
        __syncthreads();

        // Per-row normalizer dot product.
        if (tid < 128) {
            const bf16_t* qp = Qb + (size_t)(m0 + tid) * HD + h * D_OUT;
            float a = 0.0f;
            for (int m = 0; m < D_OUT; ++m) a += (float)qp[m] * ksl[m];
            normz[tid] = a;
        }
        __syncthreads();

        // GEMM: rows [wave*16, +16) x cols [0,128), K = 128.
        v8f cacc[8];
#pragma unroll
        for (int u = 0; u < 8; ++u) cacc[u] = zero8;
#pragma unroll
        for (int k0 = 0; k0 < D_OUT; k0 += 32) {
            const bf16_t* rp =
                Qb + (size_t)(m0 + wave * 16 + lm) * HD + h * D_OUT + k0;
            v16bf afr = frag_ld2(rp + khalf, rp + 16 + khalf);
#pragma unroll
            for (int u = 0; u < 8; ++u) {
                v16bf bfr = frag_ld2(&Bt[u * 16 + lm][k0 + khalf],
                                     &Bt[u * 16 + lm][k0 + 16 + khalf]);
                cacc[u] = wmma_bf16(afr, bfr, cacc[u]);
            }
        }

        float inv_den[8];
#pragma unroll
        for (int j = 0; j < 8; ++j) {
            const int rloc = wave * 16 + j + hi * 8;
            inv_den[j] = 1.0f / (normz[rloc] * inv_s + Nf);
        }
#pragma unroll
        for (int u = 0; u < 8; ++u) {
#pragma unroll
            for (int j = 0; j < 8; ++j) {
                const int row = m0 + wave * 16 + j + hi * 8;
                const int d   = u * 16 + lm;
                const float v = Vf[(size_t)row * HD + h * D_OUT + d];
                const float num = cacc[u][j] * inv_s + Nf * v;
                oacc[u][j] += num * inv_den[j];
            }
        }
    }

#pragma unroll
    for (int u = 0; u < 8; ++u) {
#pragma unroll
        for (int j = 0; j < 8; ++j) {
            const int row = m0 + wave * 16 + j + hi * 8;
            const int d   = u * 16 + lm;
            out[(size_t)row * D_OUT + d] = oacc[u][j] * 0.25f;
        }
    }
}

// ---------------------------------------------------------------------------
extern "C" void kernel_launch(void* const* d_in, const int* in_sizes, int n_in,
                              void* d_out, int out_size, void* d_ws, size_t ws_size,
                              hipStream_t stream) {
    (void)in_sizes; (void)n_in; (void)out_size; (void)ws_size;

    const float* Xq = (const float*)d_in[0];
    const float* Xs = (const float*)d_in[1];
    const float* Wq = (const float*)d_in[2];
    const float* bq = (const float*)d_in[3];
    const float* Wk = (const float*)d_in[4];
    const float* bk = (const float*)d_in[5];
    const float* Wv = (const float*)d_in[6];
    const float* bv = (const float*)d_in[7];
    float* out = (float*)d_out;

    char* ws = (char*)d_ws;
    size_t off = 0;
    bf16_t* Qb = (bf16_t*)(ws + off);  off += (size_t)N_NODES * HD * sizeof(bf16_t);
    bf16_t* Kb = (bf16_t*)(ws + off);  off += (size_t)N_NODES * HD * sizeof(bf16_t);
    float*  Vf = (float*)(ws + off);   off += (size_t)N_NODES * HD * sizeof(float);
    // Small accumulators (kept contiguous so one zero pass clears them all).
    float*  kvs    = (float*)(ws + off); off += (size_t)HEADS * D_OUT * D_OUT * sizeof(float);
    float*  ks_sum = (float*)(ws + off); off += (size_t)HD * sizeof(float);
    float*  sumsq  = (float*)(ws + off); off += 2 * sizeof(float);
    bf16_t* kvsb   = (bf16_t*)(ws + off); off += (size_t)HEADS * D_OUT * D_OUT * sizeof(bf16_t);

    const int nzero = HEADS * D_OUT * D_OUT + HD + 2;
    zero_kernel<<<(nzero + 255) / 256, 256, 0, stream>>>(kvs, nzero);

    qkv_proj_kernel<<<dim3(HD / 128, N_NODES / 128, 3), 256, 0, stream>>>(
        Xq, Xs, Wq, bq, Wk, bk, Wv, bv, Qb, Kb, Vf, sumsq, ks_sum);

    kv_outer_kernel<<<dim3(N_NODES / 2048, HEADS), 256, 0, stream>>>(Kb, Vf, kvs);

    cvt_kvs_kernel<<<(HEADS * D_OUT * D_OUT) / 256, 256, 0, stream>>>(kvs, kvsb);

    attn_out_kernel<<<N_NODES / 128, 256, 0, stream>>>(Qb, Vf, kvsb, ks_sum,
                                                       sumsq, out);
}
